// ContrastiveLoss_10376640987949
// MI455X (gfx1250) — compile-verified, matrix operand order checked by reading, not run on hardware
//
#include <hip/hip_runtime.h>

// ---------------- problem constants ----------------
constexpr int   kD    = 1024;   // feature dim
constexpr int   kB    = 4096;   // rows per view
constexpr int   kN2   = 8192;   // 2B total rows
constexpr float kInvT = 1.0f / 0.07f;
constexpr float kEps  = 1e-12f;

typedef __attribute__((ext_vector_type(16))) _Float16 v16h;
typedef __attribute__((ext_vector_type(8)))  _Float16 v8h;
typedef __attribute__((ext_vector_type(8)))  float    v8f;
typedef __attribute__((ext_vector_type(4)))  int      v4i;

union Frag {
  v16h v;
  struct { v8h lo; v8h hi; } h;
};

// CDNA5 async global->LDS staging (ASYNCcnt-tracked), if the toolchain declares it.
#if defined(__gfx1250__) && __has_builtin(__builtin_amdgcn_global_load_async_to_lds_b128) && \
    __has_builtin(__builtin_amdgcn_s_wait_asynccnt)
#define NTX_ASYNC 1
typedef __attribute__((address_space(1))) v4i gv4i;   // global int4
typedef __attribute__((address_space(3))) v4i lv4i;   // LDS int4
#else
#define NTX_ASYNC 0
#endif

// 256-thread block sum (8 waves of 32)
__device__ __forceinline__ float block_sum256(float v, float* sred) {
  #pragma unroll
  for (int m = 16; m >= 1; m >>= 1) v += __shfl_xor(v, m, 32);
  const int wid = threadIdx.x >> 5;
  if ((threadIdx.x & 31) == 0) sred[wid] = v;
  __syncthreads();
  float t = 0.f;
  #pragma unroll
  for (int i = 0; i < 8; ++i) t += sred[i];
  __syncthreads();
  return t;
}

// ---------------- kernel 1: L2-normalize rows, write f16 [8192 x 1024] ----------------
__global__ void ntx_normalize(const float* __restrict__ f1,
                              const float* __restrict__ f2,
                              _Float16* __restrict__ out) {
  __shared__ float sred[8];
  const int r = blockIdx.x;
  const float* src = (r < kB) ? (f1 + (size_t)r * kD)
                              : (f2 + (size_t)(r - kB) * kD);
  const int t = threadIdx.x;
  float v[4];
  float ss = 0.f;
  #pragma unroll
  for (int i = 0; i < 4; ++i) {
    v[i] = src[t + 256 * i];
    ss += v[i] * v[i];
  }
  const float tot   = block_sum256(ss, sred);
  const float scale = 1.0f / fmaxf(sqrtf(tot), kEps);
  _Float16* dst = out + (size_t)r * kD;
  #pragma unroll
  for (int i = 0; i < 4; ++i)
    dst[t + 256 * i] = (_Float16)(v[i] * scale);
}

// ---------------- kernel 2: positives  pos[i] = <f1n_i, f2n_i> / T ----------------
__global__ void ntx_pos(const _Float16* __restrict__ F, float* __restrict__ pos) {
  __shared__ float sred[8];
  const int i = blockIdx.x;
  const int t = threadIdx.x;
  const _Float16* a = F + (size_t)i * kD;
  const _Float16* b = F + (size_t)(i + kB) * kD;
  float acc = 0.f;
  #pragma unroll
  for (int it = 0; it < 4; ++it) {
    const int k = t + 256 * it;
    acc += (float)a[k] * (float)b[k];
  }
  const float tot = block_sum256(acc, sred);
  if (t == 0) pos[i] = tot * kInvT;
}

// Stage one 32-col x 1024-K f16 tile (64 KB) into LDS; 256 threads, 16 x b128 each.
__device__ __forceinline__ void stage_tile(const _Float16* __restrict__ F,
                                           _Float16* __restrict__ dstTile,
                                           int n0, int t) {
  #pragma unroll
  for (int it = 0; it < 16; ++it) {
    const int idx = t + it * 256;        // 0..4095 16-byte chunks
    const _Float16* src = F + (size_t)(n0 + (idx >> 7)) * kD + ((idx & 127) << 3);
    _Float16* dst = dstTile + (size_t)idx * 8;
#if NTX_ASYNC
    // flat->AS(1)/AS(3) via integer round-trip (flat low 32 bits == LDS offset)
    gv4i* g = (gv4i*)(unsigned long long)(uintptr_t)src;
    lv4i* l = (lv4i*)(unsigned)(uintptr_t)dst;
    __builtin_amdgcn_global_load_async_to_lds_b128(g, l, 0, 0);
#else
    *(uint4*)dst = *(const uint4*)src;
#endif
  }
}

__device__ __forceinline__ void wait_stage() {
#if NTX_ASYNC
  __builtin_amdgcn_s_wait_asynccnt(0);
#endif
}

// ---------------- kernel 3: fused WMMA sim + exp row-sums (diag excluded) ----------------
// Block: 256 threads = 8 waves. Each wave owns 32 rows (two 16-row subtiles) and
// sweeps all 8192 columns in 32-col steps. The 32-col B tile is double-buffered in
// LDS: tile n+1 is async-staged while the WMMAs consume tile n.
__global__ void __launch_bounds__(256)
ntx_simlse(const _Float16* __restrict__ F, float* __restrict__ rowsum) {
  __shared__ _Float16 Bs[2][32 * kD];         // 2 x 64 KB LDS tiles: [col][k]

  const int t    = threadIdx.x;
  const int lane = t & 31;
  const int wid  = t >> 5;
  const int R    = (blockIdx.x * 8 + wid) * 32;   // wave's 32 rows
  const int lr   = lane & 15;                     // lane-in-16
  const int half = lane >> 4;                     // 0: K 0-7/16-23, 1: K 8-15/24-31

  const _Float16* pA0 = F + (size_t)(R + lr) * kD;
  const _Float16* pA1 = F + (size_t)(R + 16 + lr) * kD;

  float racc[2][8];
  #pragma unroll
  for (int m = 0; m < 2; ++m)
    #pragma unroll
    for (int v = 0; v < 8; ++v) racc[m][v] = 0.f;

  stage_tile(F, &Bs[0][0], 0, t);                 // prologue: tile 0 in flight

  for (int n0 = 0; n0 < kN2; n0 += 32) {
    const int p = (n0 >> 5) & 1;

    wait_stage();          // our async loads for tile p complete
    __syncthreads();       // everyone's loads done; prior compute done (buf p^1 reusable)

    if (n0 + 32 < kN2)     // overlap: stage next tile while computing this one
      stage_tile(F, &Bs[1 - p][0], n0 + 32, t);

    const _Float16* pB0 = &Bs[p][0] + (size_t)lr * kD;
    const _Float16* pB1 = &Bs[p][0] + (size_t)(16 + lr) * kD;

    v8f c00 = {}, c01 = {}, c10 = {}, c11 = {};

    #pragma unroll 4
    for (int k0 = 0; k0 < kD; k0 += 32) {
      const int o0 = k0 + (half << 3);
      Frag a0, a1, b0, b1;
      a0.h.lo = *(const v8h*)(pA0 + o0);  a0.h.hi = *(const v8h*)(pA0 + o0 + 16);
      a1.h.lo = *(const v8h*)(pA1 + o0);  a1.h.hi = *(const v8h*)(pA1 + o0 + 16);
      b0.h.lo = *(const v8h*)(pB0 + o0);  b0.h.hi = *(const v8h*)(pB0 + o0 + 16);
      b1.h.lo = *(const v8h*)(pB1 + o0);  b1.h.hi = *(const v8h*)(pB1 + o0 + 16);

      c00 = __builtin_amdgcn_wmma_f32_16x16x32_f16(false, a0.v, false, b0.v,
                                                   (short)0, c00, false, false);
      c01 = __builtin_amdgcn_wmma_f32_16x16x32_f16(false, a0.v, false, b1.v,
                                                   (short)0, c01, false, false);
      c10 = __builtin_amdgcn_wmma_f32_16x16x32_f16(false, a1.v, false, b0.v,
                                                   (short)0, c10, false, false);
      c11 = __builtin_amdgcn_wmma_f32_16x16x32_f16(false, a1.v, false, b1.v,
                                                   (short)0, c11, false, false);
    }

    // exp() each element, accumulate per-lane row partial sums
    #pragma unroll
    for (int v = 0; v < 8; ++v) {
      racc[0][v] += __expf(c00[v] * kInvT) + __expf(c01[v] * kInvT);
      racc[1][v] += __expf(c10[v] * kInvT) + __expf(c11[v] * kInvT);
    }

    // remove the self-similarity diagonal exactly (lives in c00/c11 when n0==R)
    if (n0 == R && (lr >> 3) == half) {
      #pragma unroll
      for (int v = 0; v < 8; ++v) {
        if (v == (lr & 7)) {
          racc[0][v] -= __expf(c00[v] * kInvT);
          racc[1][v] -= __expf(c11[v] * kInvT);
        }
      }
    }
  }

  // reduce across the 16 lanes sharing each row (xor 1,2,4,8 stays in half)
  #pragma unroll
  for (int m = 0; m < 2; ++m)
    #pragma unroll
    for (int v = 0; v < 8; ++v) {
      float s = racc[m][v];
      s += __shfl_xor(s, 1, 32);
      s += __shfl_xor(s, 2, 32);
      s += __shfl_xor(s, 4, 32);
      s += __shfl_xor(s, 8, 32);
      racc[m][v] = s;
    }

  if (lr == 0) {
    // lane 0 holds rows R+0..7 / R+16..23 ; lane 16 holds rows R+8..15 / R+24..31
    #pragma unroll
    for (int v = 0; v < 8; ++v) {
      rowsum[R + 8 * half + v]      = racc[0][v];
      rowsum[R + 16 + 8 * half + v] = racc[1][v];
    }
  }
}

// ---------------- kernel 4: loss = -mean(pos - log(rowsum)) ----------------
__global__ void ntx_loss(const float* __restrict__ rowsum,
                         const float* __restrict__ pos,
                         float* __restrict__ out) {
  __shared__ float sred[8];
  float acc = 0.f;
  for (int r = threadIdx.x; r < kN2; r += 256) {
    const float p = pos[r & (kB - 1)];
    acc += p - __logf(fmaxf(rowsum[r], 1e-30f));
  }
  const float tot = block_sum256(acc, sred);
  if (threadIdx.x == 0) out[0] = -(tot / (float)kN2);
}

// ---------------- launcher ----------------
extern "C" void kernel_launch(void* const* d_in, const int* in_sizes, int n_in,
                              void* d_out, int out_size, void* d_ws, size_t ws_size,
                              hipStream_t stream) {
  (void)in_sizes; (void)n_in; (void)out_size; (void)ws_size;
  const float* f1 = (const float*)d_in[0];
  const float* f2 = (const float*)d_in[1];
  float* out = (float*)d_out;

  _Float16* Fh     = (_Float16*)d_ws;                                   // 16 MB
  float*    rowsum = (float*)((char*)d_ws + (size_t)kN2 * kD * sizeof(_Float16));
  float*    pos    = rowsum + kN2;

  ntx_normalize<<<kN2, 256, 0, stream>>>(f1, f2, Fh);
  ntx_pos<<<kB, 256, 0, stream>>>(Fh, pos);
  ntx_simlse<<<kN2 / 256, 256, 0, stream>>>(Fh, rowsum);
  ntx_loss<<<1, 256, 0, stream>>>(rowsum, pos, out);
}
